// CrossAttention_78228534329865
// MI455X (gfx1250) — matrix operand order repaired and checked by reading, not compiled, not run on hardware
//
#include <hip/hip_runtime.h>

// ---------------- problem constants ----------------
#define D_MODEL 1024
#define Hh      16
#define DHd     64
#define TQc     2048
#define TKVc    2048
#define Bc      2
#define Mrows   (TQc * Bc)       // 4096
#define NH_ROT  12               // floor(0.75 * 16)
#define NC_ROT  32               // floor(64*0.5/2)*2
#define MAX_WL_F 8192.0f

typedef __bf16 bf16_t;
typedef __attribute__((ext_vector_type(16))) __bf16 v16bf;
typedef __attribute__((ext_vector_type(8)))  float  v8f;
typedef __attribute__((ext_vector_type(4)))  unsigned int v4u;
typedef __attribute__((ext_vector_type(4)))  int v4i;

__device__ __forceinline__ v4u ld128(const bf16_t* p) {
    return *reinterpret_cast<const v4u*>(p);
}

__device__ __forceinline__ v8f wmma_bf16(v16bf a, v16bf b, v8f c) {
    return __builtin_amdgcn_wmma_f32_16x16x32_bf16(false, a, false, b, (short)0, c,
                                                   false, false);
}

// ---------------- f32 -> bf16 pack ----------------
__global__ void pack_bf16_kernel(const float* __restrict__ src,
                                 bf16_t* __restrict__ dst, int n) {
    int i = blockIdx.x * blockDim.x + threadIdx.x;
    if (i < n) dst[i] = (bf16_t)src[i];
}

// ---------------- generic WMMA GEMM:  C[m,n] = sum_k A[m,k]*W[n,k] + bias[n] --------
// A: M x K bf16 row-major, W: N x K bf16 row-major.  M=4096, N=K=1024 (compile-time).
// Per-wave output tile: 32 (M) x 64 (N); K-loop two-stage software pipelined:
// buffer0 always holds even K-steps, buffer1 odd K-steps -> no rotation copies,
// no VALU-writes-WMMA-source hazards.
// MODE 0: write f32 C (M x N row-major)
// MODE 1: write bf16 to V-transposed layout Vt[((b*H+h)*DH + d)*TKV + t]
template <int MODE>
__global__ __launch_bounds__(256)
void gemm_bf16_kernel(const bf16_t* __restrict__ A,
                      const bf16_t* __restrict__ W,
                      const float*  __restrict__ bias,
                      float*  __restrict__ Cf,
                      bf16_t* __restrict__ Cvt)
{
    constexpr int M = Mrows;
    constexpr int N = D_MODEL;
    constexpr int K = D_MODEL;
    constexpr int NSTEP = K / 32;            // 32 (even)

    const int lane = threadIdx.x & 31;
    const int wave = threadIdx.x >> 5;
    const int tile = blockIdx.x * 8 + wave;  // (M/32)*(N/64) = 2048 tiles
    const int mt = tile >> 4;                // / (N/64)
    const int nt = tile & 15;
    if (mt * 32 >= M) return;
    const int m0   = mt * 32;
    const int n0   = nt * 64;
    const int half = lane >> 4;              // 0 | 1
    const int l16  = lane & 15;

    const bf16_t* arow0 = A + (size_t)(m0 + l16) * K;
    const bf16_t* arow1 = arow0 + (size_t)16 * K;
    const bf16_t* wrow  = W + (size_t)(n0 + l16) * K + half * 8;

    // A fragment (16x32): lane<16 -> K {k..k+7, k+16..k+23}; lane>=16 -> +8
    // B fragment (32x16): lane<16 -> K {k..k+7, k+16..k+23}; lane>=16 -> +8
    auto loadStep = [&](int k, v16bf (&af)[2], v16bf (&bfr)[4]) {
        v4u* p;
        p = reinterpret_cast<v4u*>(&af[0]);
        p[0] = ld128(arow0 + k + half * 8);
        p[1] = ld128(arow0 + k + 16 + half * 8);
        p = reinterpret_cast<v4u*>(&af[1]);
        p[0] = ld128(arow1 + k + half * 8);
        p[1] = ld128(arow1 + k + 16 + half * 8);
#pragma unroll
        for (int j = 0; j < 4; ++j) {
            const bf16_t* wr = wrow + (size_t)(j * 16) * K + k;
            p = reinterpret_cast<v4u*>(&bfr[j]);
            p[0] = ld128(wr);
            p[1] = ld128(wr + 16);
        }
    };

    v8f acc0[4] = {}, acc1[4] = {};
    auto doStep = [&](const v16bf (&af)[2], const v16bf (&bfr)[4]) {
#pragma unroll
        for (int j = 0; j < 4; ++j) {
            acc0[j] = wmma_bf16(af[0], bfr[j], acc0[j]);
            acc1[j] = wmma_bf16(af[1], bfr[j], acc1[j]);
        }
    };

    v16bf a0[2], b0[4], a1[2], b1[4];
    loadStep(0, a0, b0);                     // step 0 -> buf0

#pragma unroll 1
    for (int s = 0; s < NSTEP - 2; s += 2) {
        loadStep((s + 1) * 32, a1, b1);      // odd step  -> buf1
        doStep(a0, b0);                      // consume even
        loadStep((s + 2) * 32, a0, b0);      // even step -> buf0
        doStep(a1, b1);                      // consume odd
    }
    loadStep((NSTEP - 1) * 32, a1, b1);      // last (odd) step
    doStep(a0, b0);
    doStep(a1, b1);

#pragma unroll
    for (int mi = 0; mi < 2; ++mi) {
#pragma unroll
        for (int j = 0; j < 4; ++j) {
            const int col = n0 + j * 16 + l16;
            const float bv = bias[col];
            const v8f av = mi ? acc1[j] : acc0[j];
#pragma unroll
            for (int r = 0; r < 8; ++r) {
                const int m = m0 + mi * 16 + half * 8 + r;  // C layout rows
                const float val = av[r] + bv;
                if constexpr (MODE == 0) {
                    Cf[(size_t)m * N + col] = val;
                } else {
                    const int t = m >> 1, b = m & 1;        // Bc == 2
                    const int h = col >> 6, d = col & 63;
                    Cvt[(((size_t)b * Hh + h) * DHd + d) * (size_t)TKVc + t] = (bf16_t)val;
                }
            }
        }
    }
}

// ---------------- rotary (partial) + pack to per-(b,h) row-major bf16 ----------------
// src: M x D f32 (m = t*B + b); dst: [((b*H + h)*T + t)*DH + d] bf16
__global__ void rotary_pack_kernel(const float* __restrict__ X,
                                   const float* __restrict__ pos,   // T x B
                                   bf16_t* __restrict__ Y, int T)
{
    const int idx = blockIdx.x * blockDim.x + threadIdx.x;   // one thread per channel pair
    const int total = T * Bc * (D_MODEL / 2);
    if (idx >= total) return;
    const int pair = idx % (D_MODEL / 2);
    const int m    = idx / (D_MODEL / 2);
    const int n    = pair * 2;
    const int t = m / Bc, b = m % Bc;
    const int h = n >> 6, dl = n & 63;

    float x0 = X[(size_t)m * D_MODEL + n];
    float x1 = X[(size_t)m * D_MODEL + n + 1];
    float y0 = x0, y1 = x1;
    if (h < NH_ROT && dl < NC_ROT) {
        const int i = dl >> 1;                       // rotary pair index, 0..15
        // freq_i = MAX_WL ^ ((2/nc) * i * nb/(nb-1)),  nb = nc/2 = 16
        const float expo = (2.0f / (float)NC_ROT) * ((float)i * (16.0f / 15.0f));
        const float inv  = powf(MAX_WL_F, -expo);
        const float ang  = pos[(size_t)t * Bc + b] * inv;
        const float c = cosf(ang), s = sinf(ang);
        y0 = x0 * c - x1 * s;                        // x*cos + splice(x)*sin
        y1 = x1 * c + x0 * s;
    }
    const size_t o = (((size_t)b * Hh + h) * T + t) * DHd + dl;
    Y[o]     = (bf16_t)y0;
    Y[o + 1] = (bf16_t)y1;
}

// ---------------- flash attention: one wave = one (h,b) and 16 q rows ----------------
// Qp: [b][h][TQ][64]   Kp: [b][h][TKV][64]   Vt: [b][h][64][TKV]
// mask: [TKV][B][TQ] int.  Out (bf16): [t][b][h*64+d]
__global__ __launch_bounds__(256)
void attention_kernel(const bf16_t* __restrict__ Qp,
                      const bf16_t* __restrict__ Kp,
                      const bf16_t* __restrict__ Vt,
                      const int*    __restrict__ mask,
                      bf16_t* __restrict__ Out)
{
    __shared__ __align__(16) bf16_t pl[8][16 * 32];   // per-wave 16x32 P staging

    const int lane = threadIdx.x & 31;
    const int wave = threadIdx.x >> 5;
    const int tile = blockIdx.x * 8 + wave;           // H*B*(TQ/16) = 4096 total
    const int qt = tile & ((TQc / 16) - 1);
    const int hb = tile >> 7;
    const int h  = hb & (Hh - 1);
    const int b  = hb >> 4;
    const int t0 = qt * 16;
    const int half = lane >> 4;
    const int l16  = lane & 15;

    const bf16_t* qbase = Qp + (((size_t)b * Hh + h) * TQc + t0) * DHd;
    const bf16_t* kbase = Kp + (((size_t)b * Hh + h) * TKVc) * DHd;
    const bf16_t* vbase = Vt + (((size_t)b * Hh + h) * DHd) * (size_t)TKVc;

    // Q A-fragments for the two K-dim halves (d 0..31 and 32..63), kept resident.
    v16bf qa[2];
    {
        const bf16_t* qrow = qbase + (size_t)l16 * DHd;
#pragma unroll
        for (int db = 0; db < 2; ++db) {
            v4u* p = reinterpret_cast<v4u*>(&qa[db]);
            p[0] = ld128(qrow + db * 32 + half * 8);
            p[1] = ld128(qrow + db * 32 + 16 + half * 8);
        }
    }

    v8f o[4] = {};
    v8f mrow, lrow;
#pragma unroll
    for (int r = 0; r < 8; ++r) { mrow[r] = -3.0e30f; lrow[r] = 0.0f; }

    bf16_t* plw = pl[wave];

#pragma unroll 1
    for (int kv0 = 0; kv0 < TKVc; kv0 += 32) {
        // prefetch next iteration's K rows into cache
        __builtin_prefetch(kbase + (size_t)(kv0 + 32 + l16) * DHd, 0, 1);

        // ---- load both kv tiles' K fragments in one clause ----
        v16bf kb[2][2];                               // [st][db]
#pragma unroll
        for (int st = 0; st < 2; ++st) {
            const bf16_t* krow = kbase + (size_t)(kv0 + st * 16 + l16) * DHd;
#pragma unroll
            for (int db = 0; db < 2; ++db) {
                v4u* p = reinterpret_cast<v4u*>(&kb[st][db]);
                p[0] = ld128(krow + db * 32 + half * 8);
                p[1] = ld128(krow + db * 32 + half * 8 + 16);
            }
        }

        // ---- S = Q @ K^T: interleave the two accumulators (dependency distance 2) ----
        v8f sacc[2] = {};
        sacc[0] = wmma_bf16(qa[0], kb[0][0], sacc[0]);
        sacc[1] = wmma_bf16(qa[0], kb[1][0], sacc[1]);
        sacc[0] = wmma_bf16(qa[1], kb[0][1], sacc[0]);
        sacc[1] = wmma_bf16(qa[1], kb[1][1], sacc[1]);

        // ---- hoist V fragment loads: independent of softmax, latency hidden ----
        v16bf vb[4];
#pragma unroll
        for (int j = 0; j < 4; ++j) {
            const bf16_t* vrow = vbase + (size_t)(j * 16 + l16) * TKVc + kv0 + half * 8;
            v4u* p = reinterpret_cast<v4u*>(&vb[j]);
            p[0] = ld128(vrow);
            p[1] = ld128(vrow + 16);
        }

        // ---- scale + mask (C layout: lane owns kv col kv0+st*16+l16, rows half*8+r) ----
        v8f s[2];
#pragma unroll
        for (int st = 0; st < 2; ++st) {
            const int col = kv0 + st * 16 + l16;
            const int* mp = mask + ((size_t)col * Bc + b) * TQc + t0 + half * 8;
            const v4i mk0 = *reinterpret_cast<const v4i*>(mp);
            const v4i mk1 = *reinterpret_cast<const v4i*>(mp + 4);
#pragma unroll
            for (int r = 0; r < 8; ++r) {
                const float v = sacc[st][r] * 0.125f;     // 1/sqrt(64)
                const int mv = (r < 4) ? mk0[r] : mk1[r - 4];
                s[st][r] = (mv > 0) ? v : -1.0e30f;
            }
        }

        // ---- online softmax (rows live within each 16-lane half) ----
        v8f mnew, alpha, psum;
#pragma unroll
        for (int r = 0; r < 8; ++r) {
            float mx = fmaxf(s[0][r], s[1][r]);
#pragma unroll
            for (int off = 8; off >= 1; off >>= 1)
                mx = fmaxf(mx, __shfl_xor(mx, off, 32));
            mnew[r] = fmaxf(mrow[r], mx);
        }
#pragma unroll
        for (int r = 0; r < 8; ++r) alpha[r] = __expf(mrow[r] - mnew[r]);
#pragma unroll
        for (int r = 0; r < 8; ++r) {
            const float p0 = __expf(s[0][r] - mnew[r]);
            const float p1 = __expf(s[1][r] - mnew[r]);
            s[0][r] = p0; s[1][r] = p1;
            float ps = p0 + p1;
#pragma unroll
            for (int off = 8; off >= 1; off >>= 1)
                ps += __shfl_xor(ps, off, 32);
            psum[r] = ps;
        }
#pragma unroll
        for (int r = 0; r < 8; ++r) {
            lrow[r] = lrow[r] * alpha[r] + psum[r];
            mrow[r] = mnew[r];
        }
#pragma unroll
        for (int j = 0; j < 4; ++j)
#pragma unroll
            for (int r = 0; r < 8; ++r) o[j][r] *= alpha[r];

        // ---- P (C layout) -> LDS -> A-fragment layout ----
#pragma unroll
        for (int st = 0; st < 2; ++st) {
            const int colL = st * 16 + l16;
#pragma unroll
            for (int r = 0; r < 8; ++r)
                plw[(half * 8 + r) * 32 + colL] = (bf16_t)s[st][r];
        }
        asm volatile("s_wait_dscnt 0" ::: "memory");
        v16bf pf;
        {
            const bf16_t* prow = plw + l16 * 32;
            v4u* p = reinterpret_cast<v4u*>(&pf);
            p[0] = *reinterpret_cast<const v4u*>(prow + half * 8);
            p[1] = *reinterpret_cast<const v4u*>(prow + 16 + half * 8);
        }
        asm volatile("" ::: "memory");   // keep next iteration's LDS stores after these loads

        // ---- O += P @ V ----
#pragma unroll
        for (int j = 0; j < 4; ++j)
            o[j] = wmma_bf16(pf, vb[j], o[j]);
    }

    // ---- finalize: divide by l, store bf16 into (t, b, h*64+d) ----
#pragma unroll
    for (int r = 0; r < 8; ++r) lrow[r] = (lrow[r] > 0.0f) ? (1.0f / lrow[r]) : 0.0f;
#pragma unroll
    for (int j = 0; j < 4; ++j) {
        const int dcol = h * DHd + j * 16 + l16;
#pragma unroll
        for (int r = 0; r < 8; ++r) {
            const int t = t0 + half * 8 + r;
            Out[((size_t)t * Bc + b) * D_MODEL + dcol] = (bf16_t)(o[j][r] * lrow[r]);
        }
    }
}

// ---------------- host-side launcher ----------------
extern "C" void kernel_launch(void* const* d_in, const int* in_sizes, int n_in,
                              void* d_out, int out_size, void* d_ws, size_t ws_size,
                              hipStream_t stream) {
    const float* inputs_q  = (const float*)d_in[0];
    const float* inputs_kv = (const float*)d_in[1];
    const int*   mask      = (const int*)  d_in[2];
    const float* q_pos     = (const float*)d_in[3];
    const float* kv_pos    = (const float*)d_in[4];
    const float* Wq = (const float*)d_in[5];
    const float* bq = (const float*)d_in[6];
    const float* Wk = (const float*)d_in[7];
    const float* bk = (const float*)d_in[8];
    const float* Wv = (const float*)d_in[9];
    const float* bv = (const float*)d_in[10];
    const float* Wo = (const float*)d_in[11];
    const float* bo = (const float*)d_in[12];
    float* out = (float*)d_out;

    const size_t MD = (size_t)Mrows * D_MODEL;   // 4096*1024
    const size_t DD = (size_t)D_MODEL * D_MODEL; // 1024*1024

    char* ws = (char*)d_ws;
    bf16_t* xq     = (bf16_t*)ws; ws += MD * 2;  // reused as 'summed' after attention
    bf16_t* xkv    = (bf16_t*)ws; ws += MD * 2;
    bf16_t* wqb    = (bf16_t*)ws; ws += DD * 2;
    bf16_t* wkb    = (bf16_t*)ws; ws += DD * 2;
    bf16_t* wvb    = (bf16_t*)ws; ws += DD * 2;
    bf16_t* wob    = (bf16_t*)ws; ws += DD * 2;
    float*  tmpf   = (float*) ws; ws += MD * 4;  // reused for q then k (f32, pre-rotary)
    bf16_t* rq     = (bf16_t*)ws; ws += MD * 2;
    bf16_t* rk     = (bf16_t*)ws; ws += MD * 2;
    bf16_t* vt     = (bf16_t*)ws; ws += MD * 2;
    bf16_t* summed = xq;                          // alias: xq dead after Q GEMM

    const int PT = 256;
    // 1) pack activations + weights to bf16
    pack_bf16_kernel<<<(int)(MD / PT), PT, 0, stream>>>(inputs_q,  xq,  (int)MD);
    pack_bf16_kernel<<<(int)(MD / PT), PT, 0, stream>>>(inputs_kv, xkv, (int)MD);
    pack_bf16_kernel<<<(int)(DD / PT), PT, 0, stream>>>(Wq, wqb, (int)DD);
    pack_bf16_kernel<<<(int)(DD / PT), PT, 0, stream>>>(Wk, wkb, (int)DD);
    pack_bf16_kernel<<<(int)(DD / PT), PT, 0, stream>>>(Wv, wvb, (int)DD);
    pack_bf16_kernel<<<(int)(DD / PT), PT, 0, stream>>>(Wo, wob, (int)DD);

    const int gemm_blocks = (Mrows / 32) * (D_MODEL / 64) / 8;  // 256
    const int rot_blocks  = (int)((MD / 2) / PT);               // one thread per pair

    // 2) Q = xq @ Wq^T + bq  -> f32, then rotary -> rq (bf16, per-(b,h) layout)
    gemm_bf16_kernel<0><<<gemm_blocks, 256, 0, stream>>>(xq, wqb, bq, tmpf, nullptr);
    rotary_pack_kernel<<<rot_blocks, PT, 0, stream>>>(tmpf, q_pos, rq, TQc);

    // 3) K -> f32 -> rotary -> rk
    gemm_bf16_kernel<0><<<gemm_blocks, 256, 0, stream>>>(xkv, wkb, bk, tmpf, nullptr);
    rotary_pack_kernel<<<rot_blocks, PT, 0, stream>>>(tmpf, kv_pos, rk, TKVc);

    // 4) V -> bf16 directly in transposed [b][h][d][kv] layout
    gemm_bf16_kernel<1><<<gemm_blocks, 256, 0, stream>>>(xkv, wvb, bv, nullptr, vt);

    // 5) flash attention -> summed (bf16, (t,b,D) row-major); reuses xq storage
    const int attn_blocks = (Hh * Bc * (TQc / 16)) / 8;         // 512
    attention_kernel<<<attn_blocks, 256, 0, stream>>>(rq, rk, vt, mask, summed);

    // 6) out = summed @ Wo^T + bo  -> f32 to d_out
    gemm_bf16_kernel<0><<<gemm_blocks, 256, 0, stream>>>(summed, wob, bo, out, nullptr);
}
